// BiMambaBlock_8134668058873
// MI455X (gfx1250) — matrix operand order repaired and checked by reading, not compiled, not run on hardware
//
#include <hip/hip_runtime.h>
#include <hip/hip_bf16.h>
#include <cmath>

// ---------------- problem constants ----------------
#define BATCH  4
#define SEQ    1024
#define DMODEL 512
#define DSTATE 16
#define DCONVK 4
#define DI     1024           // EXPAND * D_MODEL
#define DTR    32             // DT_RANK
#define MROWS  (BATCH * SEQ)  // 4096 token rows

// ---------------- vector types ----------------
typedef __attribute__((ext_vector_type(16))) __bf16 v16bf;
typedef __attribute__((ext_vector_type(8)))  __bf16 v8bf;
typedef __attribute__((ext_vector_type(8)))  float  v8f;

// ---------------- GEMM tiling ----------------
#define BM 64
#define BN 128
#define BK 32
#define LDT 40   // bf16 element stride for LDS tiles (pad, 16B aligned rows)

__device__ __forceinline__ float act_apply(float v, int act) {
  if (act == 1) {                       // softplus
    return (v > 20.f) ? v : log1pf(__expf(v));
  }
  if (act == 2) {                       // exact GELU
    return 0.5f * v * (1.f + erff(v * 0.70710678118654752f));
  }
  return v;
}

// one-time f32 -> bf16 weight conversion (halves weight tile traffic forever after)
__global__ __launch_bounds__(256) void f32_to_bf16_kernel(
    const float* __restrict__ s, __bf16* __restrict__ d, int n) {
  const int i = (blockIdx.x * 256 + threadIdx.x) * 4;
  if (i < n) {
#pragma unroll
    for (int j = 0; j < 4; ++j) d[i + j] = (__bf16)s[i + j];
  }
}

// C[M,N] = act( A[M,K](f32) * W[N,K](bf16)^T + bias[N] )
// A tile: global f32 -> regs -> bf16 -> LDS (VALU convert, overlapped)
// W tile: GLOBAL_LOAD_ASYNC_TO_LDS_B128 raw bf16 copy (ASYNCcnt), LDS double buffered
__global__ __launch_bounds__(256) void gemm_wmma_bf16(
    const float* __restrict__ A, int lda,
    const __bf16* __restrict__ W, int ldw,
    const float* __restrict__ bias,
    float* __restrict__ C, int ldc,
    int M, int N, int K, int act)
{
  __shared__ alignas(16) __bf16 As[2][BM * LDT];   // [64][32]+pad, double buffered
  __shared__ alignas(16) __bf16 Ws[2][BN * LDT];   // [128][32]+pad, double buffered

  const int tid   = threadIdx.x;
  const int lane  = tid & 31;
  const int wave  = tid >> 5;       // 0..7
  const int waveM = wave & 1;       // 2 waves over M
  const int waveN = wave >> 1;      // 4 waves over N
  const int r     = lane & 15;
  const int half  = lane >> 4;
  const int tileM = blockIdx.y * BM;
  const int tileN = blockIdx.x * BN;

  v8f acc[2][2];
#pragma unroll
  for (int i = 0; i < 2; ++i)
#pragma unroll
    for (int j = 0; j < 2; ++j)
#pragma unroll
      for (int e = 0; e < 8; ++e) acc[i][j][e] = 0.f;

  const int arow = tid >> 2;          // 0..63
  const int acol = (tid & 3) * 8;     // 0,8,16,24

  auto issue_tile = [&](int k0, int buf) {
    // ---- A tile: f32 -> bf16 via VALU, 8 elems/thread, one ds_store_b128 ----
    {
      const float* src = A + (size_t)(tileM + arow) * lda + k0 + acol;
      v8bf t;
#pragma unroll
      for (int i = 0; i < 8; ++i) t[i] = (__bf16)src[i];
      *(v8bf*)&As[buf][arow * LDT + acol] = t;
    }
    // ---- W tile: async bf16 copy straight into LDS, 2 x b128 per thread ----
#pragma unroll
    for (int rr = 0; rr < 2; ++rr) {
      const int cid  = rr * 256 + tid;     // 512 chunks of 8 bf16 (16B)
      const int wrow = cid >> 2;           // 0..127
      const int wcol = (cid & 3) * 8;      // 0,8,16,24
      const int gn   = tileN + wrow;
      if (gn < N) {
        // low 32 bits of the flat shared address == LDS byte offset
        const unsigned ldsa = (unsigned)(unsigned long long)&Ws[buf][wrow * LDT + wcol];
        const __bf16* ga = W + (size_t)gn * ldw + k0 + wcol;
        asm volatile("global_load_async_to_lds_b128 %0, %1, off"
                     :: "v"(ldsa), "v"(ga) : "memory");
      }
    }
  };

  int buf = 0;
  issue_tile(0, 0);
  asm volatile("s_wait_asynccnt 0x0" ::: "memory");
  __syncthreads();

  for (int k0 = 0; k0 < K; k0 += BK) {
    if (k0 + BK < K) issue_tile(k0 + BK, buf ^ 1);   // stream next tile while computing

    // ---- WMMA compute: 2x2 fragments of 16x16 per wave ----
#pragma unroll
    for (int mi = 0; mi < 2; ++mi) {
      const int am = waveM * 32 + mi * 16 + r;
      // A frag: lane(r,half) holds K = half*16 .. half*16+15
      v8bf alo = *(const v8bf*)&As[buf][am * LDT + half * 16];
      v8bf ahi = *(const v8bf*)&As[buf][am * LDT + half * 16 + 8];
      v16bf afrag = __builtin_shufflevector(alo, ahi,
          0,1,2,3,4,5,6,7,8,9,10,11,12,13,14,15);
#pragma unroll
      for (int nj = 0; nj < 2; ++nj) {
        const int wn = waveN * 32 + nj * 16 + r;
        // B frag: lane col = r, K = half*16 + i  ->  Ws[wn][half*16 .. +15]
        v8bf blo = *(const v8bf*)&Ws[buf][wn * LDT + half * 16];
        v8bf bhi = *(const v8bf*)&Ws[buf][wn * LDT + half * 16 + 8];
        v16bf bfrag = __builtin_shufflevector(blo, bhi,
            0,1,2,3,4,5,6,7,8,9,10,11,12,13,14,15);
        acc[mi][nj] = __builtin_amdgcn_wmma_f32_16x16x32_bf16(
            false, afrag, false, bfrag, (short)0, acc[mi][nj], false, false);
      }
    }

    asm volatile("s_wait_asynccnt 0x0" ::: "memory");  // this wave's async fills done
    __syncthreads();                                   // all waves' fills visible
    buf ^= 1;
  }

  // ---- epilogue: C/D layout -> row = half*8 + i, col = r ----
#pragma unroll
  for (int mi = 0; mi < 2; ++mi) {
#pragma unroll
    for (int nj = 0; nj < 2; ++nj) {
      const int col = tileN + waveN * 32 + nj * 16 + r;
      if (col < N) {
        const float bv = bias ? bias[col] : 0.f;
#pragma unroll
        for (int i = 0; i < 8; ++i) {
          const int row = tileM + waveM * 32 + mi * 16 + half * 8 + i;
          float v = acc[mi][nj][i] + bv;
          C[(size_t)row * ldc + col] = act_apply(v, act);
        }
      }
    }
  }
}

// ---------------- depthwise causal conv (both directions) + SiLU ----------------
__global__ __launch_bounds__(256) void conv_silu_kernel(
    const float* __restrict__ xz, const float* __restrict__ cw,
    const float* __restrict__ cb, float* __restrict__ xc)
{
  const int id = blockIdx.x * blockDim.x + threadIdx.x;   // 2*B*L*DI = 2^23
  const int c = id & (DI - 1);
  const int p = (id >> 10) & (SEQ - 1);
  const int b = (id >> 20) & (BATCH - 1);
  const int d = id >> 22;

  const float* xm = xz + ((size_t)d * MROWS) * (2 * DI);
  float acc = cb[d * DI + c];
#pragma unroll
  for (int k = 0; k < DCONVK; ++k) {
    const int q = (d == 0) ? (p - (DCONVK - 1) + k) : (p + (DCONVK - 1) - k);
    if (q >= 0 && q < SEQ)
      acc += cw[((size_t)d * DI + c) * DCONVK + k] *
             xm[((size_t)b * SEQ + q) * (2 * DI) + c];
  }
  acc = acc / (1.f + __expf(-acc));   // SiLU
  xc[((size_t)d * MROWS + (size_t)b * SEQ + p) * DI + c] = acc;
}

// ---------------- S6 selective scan (sequential over L, parallel over d,b,ch) -------
__global__ __launch_bounds__(256) void ssm_scan_kernel(
    const float* __restrict__ delta, const float* __restrict__ xdbl,
    const float* __restrict__ xz, const float* __restrict__ A_log,
    const float* __restrict__ Dp, float* __restrict__ xc_y)
{
  const int d  = blockIdx.z;
  const int b  = blockIdx.y;
  const int ch = blockIdx.x * blockDim.x + threadIdx.x;   // 0..DI-1

  float Arow[DSTATE], h[DSTATE];
#pragma unroll
  for (int s = 0; s < DSTATE; ++s) {
    Arow[s] = -__expf(A_log[((size_t)d * DI + ch) * DSTATE + s]);
    h[s] = 0.f;
  }
  const float Dv = Dp[d * DI + ch];

  __shared__ float Bs[DSTATE];
  __shared__ float Cs[DSTATE];

  const float* xdb = xdbl  + ((size_t)d * MROWS + (size_t)b * SEQ) * 64;
  const float* del = delta + ((size_t)d * MROWS + (size_t)b * SEQ) * DI;
  const float* zp  = xz    + ((size_t)d * MROWS + (size_t)b * SEQ) * (2 * DI) + DI;
  float*       xy  = xc_y  + ((size_t)d * MROWS + (size_t)b * SEQ) * DI;

  for (int t = 0; t < SEQ; ++t) {
    const int p = (d == 0) ? t : (SEQ - 1 - t);
    if (threadIdx.x < 2 * DSTATE) {
      // xdbl cols 32..47 = B, 48..63 = C
      const float v = xdb[(size_t)p * 64 + DTR + threadIdx.x];
      if (threadIdx.x < DSTATE) Bs[threadIdx.x] = v;
      else                      Cs[threadIdx.x - DSTATE] = v;
    }
    __syncthreads();

    const float dv = del[(size_t)p * DI + ch];
    const float xv = xy[(size_t)p * DI + ch];
    float yacc = 0.f;
#pragma unroll
    for (int s = 0; s < DSTATE; ++s) {
      const float dA = __expf(dv * Arow[s]);
      h[s] = dA * h[s] + dv * Bs[s] * xv;
      yacc += h[s] * Cs[s];
    }
    float yv = yacc + Dv * xv;
    const float zv = zp[(size_t)p * (2 * DI) + ch];
    yv *= zv / (1.f + __expf(-zv));        // y * silu(z)
    xy[(size_t)p * DI + ch] = yv;          // in-place (same element this thread read)
    __syncthreads();
  }
}

// ---------------- residual + LayerNorm (row length 512, block per row) ----------------
__device__ __forceinline__ float block_sum(float v, float* sbuf) {
  const int tid = threadIdx.x;
  sbuf[tid] = v;
  __syncthreads();
  for (int s = 128; s > 0; s >>= 1) {
    if (tid < s) sbuf[tid] += sbuf[tid + s];
    __syncthreads();
  }
  const float r = sbuf[0];
  __syncthreads();
  return r;
}

__global__ __launch_bounds__(256) void resid_ln_kernel(
    const float* __restrict__ a, const float* __restrict__ b,
    const float* __restrict__ c, const float* __restrict__ g,
    const float* __restrict__ beta, float* __restrict__ out)
{
  __shared__ float sbuf[256];
  const int tid = threadIdx.x;
  const size_t base = (size_t)blockIdx.x * DMODEL;

  float v0 = a[base + tid];
  float v1 = a[base + tid + 256];
  if (b) { v0 += b[base + tid]; v1 += b[base + tid + 256]; }
  if (c) { v0 += c[base + tid]; v1 += c[base + tid + 256]; }

  const float mean = block_sum(v0 + v1, sbuf) * (1.f / DMODEL);
  const float d0 = v0 - mean, d1 = v1 - mean;
  const float var = block_sum(d0 * d0 + d1 * d1, sbuf) * (1.f / DMODEL);
  const float rstd = rsqrtf(var + 1e-5f);

  out[base + tid]       = d0 * rstd * g[tid]       + beta[tid];
  out[base + tid + 256] = d1 * rstd * g[tid + 256] + beta[tid + 256];
}

// ---------------- host launch ----------------
extern "C" void kernel_launch(void* const* d_in, const int* in_sizes, int n_in,
                              void* d_out, int out_size, void* d_ws, size_t ws_size,
                              hipStream_t stream) {
  (void)in_sizes; (void)n_in; (void)out_size; (void)ws_size;
  const float* x      = (const float*)d_in[0];   // [B,L,512]
  const float* Wi     = (const float*)d_in[1];   // [2][2048][512]
  const float* cw     = (const float*)d_in[2];   // [2][1024][4]
  const float* cb     = (const float*)d_in[3];   // [2][1024]
  const float* Wx     = (const float*)d_in[4];   // [2][64][1024]
  const float* Wdt    = (const float*)d_in[5];   // [2][1024][32]
  const float* bdt    = (const float*)d_in[6];   // [2][1024]
  const float* A_log  = (const float*)d_in[7];   // [2][1024][16]
  const float* Dp     = (const float*)d_in[8];   // [2][1024]
  const float* Wo     = (const float*)d_in[9];   // [2][512][1024]
  const float* ng     = (const float*)d_in[10];
  const float* nb     = (const float*)d_in[11];
  const float* flg    = (const float*)d_in[12];
  const float* flb    = (const float*)d_in[13];
  const float* W1     = (const float*)d_in[14];  // [1024][512]
  const float* b1     = (const float*)d_in[15];
  const float* W2     = (const float*)d_in[16];  // [512][1024]
  const float* b2     = (const float*)d_in[17];
  const float* fng    = (const float*)d_in[18];
  const float* fnb    = (const float*)d_in[19];
  float* out = (float*)d_out;

  // f32 workspace layout; later-phase buffers alias freed regions
  float* ws    = (float*)d_ws;
  float* XZ    = ws;                                  // [2][4096][2048]
  float* XC    = XZ   + (size_t)2 * MROWS * 2 * DI;   // [2][4096][1024]
  float* XDBL  = XC   + (size_t)2 * MROWS * DI;       // [2][4096][64]
  float* DELTA = XDBL + (size_t)2 * MROWS * 64;       // [2][4096][1024]
  float* O0  = DELTA;                                 // alias (free after scan)
  float* O1  = O0 + (size_t)MROWS * DMODEL;
  float* X2  = O1 + (size_t)MROWS * DMODEL;
  float* HLN = X2 + (size_t)MROWS * DMODEL;
  float* H1  = XZ;                                    // alias (free after scan)
  float* H2  = H1 + (size_t)MROWS * DI;

  // bf16 weight copies, placed after the f32 region (~8.4 MB)
  __bf16* WB   = (__bf16*)(DELTA + (size_t)2 * MROWS * DI);
  __bf16* WiB  = WB;                                   // 2*2048*512
  __bf16* WxB  = WiB  + (size_t)2 * 2 * DI * DMODEL;   // 2*64*1024
  __bf16* WdtB = WxB  + (size_t)2 * 64 * DI;           // 2*1024*32
  __bf16* WoB  = WdtB + (size_t)2 * DI * DTR;          // 2*512*1024
  __bf16* W1B  = WoB  + (size_t)2 * DMODEL * DI;       // 1024*512
  __bf16* W2B  = W1B  + (size_t)DI * DMODEL;           // 512*1024

  const dim3 blk(256);
  auto cvt = [&](const float* s, __bf16* d, int n) {
    f32_to_bf16_kernel<<<(n / 4 + 255) / 256, blk, 0, stream>>>(s, d, n);
  };
  cvt(Wi,  WiB,  2 * 2 * DI * DMODEL);
  cvt(Wx,  WxB,  2 * 64 * DI);
  cvt(Wdt, WdtB, 2 * DI * DTR);
  cvt(Wo,  WoB,  2 * DMODEL * DI);
  cvt(W1,  W1B,  DI * DMODEL);
  cvt(W2,  W2B,  DMODEL * DI);

  auto gemm_grid = [](int N) { return dim3((unsigned)((N + BN - 1) / BN), MROWS / BM); };

  // 1) in_proj: xz[d] = x @ Wi[d]^T
  for (int d = 0; d < 2; ++d)
    gemm_wmma_bf16<<<gemm_grid(2 * DI), blk, 0, stream>>>(
        x, DMODEL, WiB + (size_t)d * 2 * DI * DMODEL, DMODEL, nullptr,
        XZ + (size_t)d * MROWS * 2 * DI, 2 * DI, MROWS, 2 * DI, DMODEL, 0);

  // 2) depthwise conv + silu (both directions)
  conv_silu_kernel<<<(2 * MROWS * DI) / 256, blk, 0, stream>>>(XZ, cw, cb, XC);

  // 3) x_proj: xdbl[d] = xc[d] @ Wx[d]^T   (N=64)
  for (int d = 0; d < 2; ++d)
    gemm_wmma_bf16<<<gemm_grid(DTR + 2 * DSTATE), blk, 0, stream>>>(
        XC + (size_t)d * MROWS * DI, DI, WxB + (size_t)d * 64 * DI, DI, nullptr,
        XDBL + (size_t)d * MROWS * 64, 64, MROWS, DTR + 2 * DSTATE, DI, 0);

  // 4) dt_proj + softplus: delta[d] = softplus(dt @ Wdt[d]^T + bdt[d])  (K=32, lda=64)
  for (int d = 0; d < 2; ++d)
    gemm_wmma_bf16<<<gemm_grid(DI), blk, 0, stream>>>(
        XDBL + (size_t)d * MROWS * 64, 64, WdtB + (size_t)d * DI * DTR, DTR,
        bdt + (size_t)d * DI, DELTA + (size_t)d * MROWS * DI, DI, MROWS, DI, DTR, 1);

  // 5) selective scan + gating (y written in place over XC)
  ssm_scan_kernel<<<dim3(DI / 256, BATCH, 2), blk, 0, stream>>>(
      DELTA, XDBL, XZ, A_log, Dp, XC);

  // 6) out_proj per direction
  for (int d = 0; d < 2; ++d)
    gemm_wmma_bf16<<<gemm_grid(DMODEL), blk, 0, stream>>>(
        XC + (size_t)d * MROWS * DI, DI, WoB + (size_t)d * DMODEL * DI, DI, nullptr,
        (d == 0 ? O0 : O1), DMODEL, MROWS, DMODEL, DI, 0);

  // 7) x2 = LN(x + fwd + bwd);  hln = LN2(x2)
  resid_ln_kernel<<<MROWS, blk, 0, stream>>>(x, O0, O1, ng, nb, X2);
  resid_ln_kernel<<<MROWS, blk, 0, stream>>>(X2, nullptr, nullptr, flg, flb, HLN);

  // 8) FFN: H1 = gelu(hln @ W1^T + b1);  H2 = H1 @ W2^T + b2
  gemm_wmma_bf16<<<gemm_grid(DI), blk, 0, stream>>>(
      HLN, DMODEL, W1B, DMODEL, b1, H1, DI, MROWS, DI, DMODEL, 2);
  gemm_wmma_bf16<<<gemm_grid(DMODEL), blk, 0, stream>>>(
      H1, DI, W2B, DI, b2, H2, DMODEL, MROWS, DMODEL, DI, 0);

  // 9) out = LN(x2 + H2)
  resid_ln_kernel<<<MROWS, blk, 0, stream>>>(X2, H2, nullptr, fng, fnb, out);
}
